// GCN_4741643895756
// MI455X (gfx1250) — matrix operand order repaired and checked by reading, not compiled, not run on hardware
//
#include <hip/hip_runtime.h>
#include <hip/hip_bf16.h>

#define N_NODES 10000
#define N_EDGES 640000
#define E_TOTAL (N_EDGES + N_NODES)   // self loops appended
#define D 128
#define LDS_STRIDE (D + 8)            // halves; 272B rows: 16B-aligned, bank-skewed

typedef _Float16 v16h __attribute__((ext_vector_type(16)));
typedef float    v8f  __attribute__((ext_vector_type(8)));

// ---------------------------------------------------------------------------
// Utility: zero a float buffer (workspace is poisoned by the harness).
// ---------------------------------------------------------------------------
__global__ void zero_kernel(float* __restrict__ p, int n) {
    int i = blockIdx.x * blockDim.x + threadIdx.x;
    if (i < n) p[i] = 0.0f;
}

// ---------------------------------------------------------------------------
// deg[dst] += 1 over all edges incl. self loops
// ---------------------------------------------------------------------------
__global__ void degree_kernel(const long long* __restrict__ dst,
                              float* __restrict__ deg) {
    int e = blockIdx.x * blockDim.x + threadIdx.x;
    if (e >= E_TOTAL) return;
    int d = (e < N_EDGES) ? (int)dst[e] : (e - N_EDGES);
    __hip_atomic_fetch_add(&deg[d], 1.0f, __ATOMIC_RELAXED, __HIP_MEMORY_SCOPE_AGENT);
}

__global__ void dinv_kernel(const float* __restrict__ deg,
                            float* __restrict__ dinv) {
    int i = blockIdx.x * blockDim.x + threadIdx.x;
    if (i >= N_NODES) return;
    float dg = deg[i];
    dinv[i] = (dg > 0.0f) ? rsqrtf(dg) : 0.0f;
}

__global__ void norm_kernel(const long long* __restrict__ src,
                            const long long* __restrict__ dst,
                            const float* __restrict__ dinv,
                            float* __restrict__ nrm) {
    int e = blockIdx.x * blockDim.x + threadIdx.x;
    if (e >= E_TOTAL) return;
    int s, d;
    if (e < N_EDGES) { s = (int)src[e]; d = (int)dst[e]; }
    else             { s = e - N_EDGES; d = s; }
    nrm[e] = dinv[s] * dinv[d];
}

// ---------------------------------------------------------------------------
// WMMA GEMM: Y[10000,128] = H[10000,128] @ W[128,128]  (f16 mul, f32 acc)
// Block = 256 threads = 8 waves. Block b owns M-tile b (16 rows), wave w owns
// N-tile w (16 cols). A tile is staged row-major; W is staged TRANSPOSED
// (Wt[n][k]) so both A- and B-fragment reads are K-contiguous -> ds_load_b128.
// Each wave issues 4x v_wmma_f32_16x16x32_f16 over K=128.
// ---------------------------------------------------------------------------
__global__ __launch_bounds__(256) void gemm_wmma_f16(const float* __restrict__ H,
                                                     const float* __restrict__ W,
                                                     float* __restrict__ Y) {
    __shared__ _Float16 Wt[D * LDS_STRIDE];   // ~34 KB, Wt[n*S + k] = W[k][n]
    __shared__ _Float16 Al[16 * LDS_STRIDE];  // ~4.3 KB

    const int tid = threadIdx.x;
    const int m0  = blockIdx.x * 16;

    // Stage W transposed: coalesced global read, permuted LDS write.
    for (int i = tid; i < D * D; i += 256) {
        int k = i >> 7, n = i & (D - 1);
        Wt[n * LDS_STRIDE + k] = (_Float16)W[i];
    }
    // Stage the 16-row A tile (row-major, K-contiguous).
    for (int i = tid; i < 16 * D; i += 256) {
        int r = i >> 7, c = i & (D - 1);
        Al[r * LDS_STRIDE + c] = (_Float16)H[(m0 + r) * D + c];
    }
    __syncthreads();

    const int wave   = tid >> 5;
    const int lane   = tid & 31;
    const int n0     = wave * 16;
    const int row    = lane & 15;          // A fragment: M = lane & 15
    const int khalfA = (lane >> 4) << 4;   // A: +16 K-offset for upper lanes
    const int kbB    = (lane >> 4) << 4;   // B: +16 K-offset for upper lanes
    const int col    = n0 + (lane & 15);   // B/C/D fragment: N = lane & 15

    v8f acc = {};
#pragma unroll
    for (int k0 = 0; k0 < D; k0 += 32) {
        v16h a, b;
#pragma unroll
        for (int j = 0; j < 16; ++j) {
            // A fragment, contiguous k order used here (same as B): elem j -> K = khalf + j
            int ka = j + khalfA;
            a[j] = Al[row * LDS_STRIDE + k0 + ka];
        }
#pragma unroll
        for (int j = 0; j < 16; ++j) {
            // B fragment, contiguous k order used here (same as A): lane holds col N, K = kbB + j
            b[j] = Wt[col * LDS_STRIDE + k0 + kbB + j];
        }
        acc = __builtin_amdgcn_wmma_f32_16x16x32_f16(
            /*neg_a=*/false, a, /*neg_b=*/false, b,
            /*c_mod=*/(short)0, acc, /*reuse_a=*/false, /*reuse_b=*/false);
    }

    // C/D layout: VGPR r -> M = r + 8*(lane>=16), N = lane & 15
    const int mrow = m0 + ((lane >> 4) << 3);
#pragma unroll
    for (int r = 0; r < 8; ++r) {
        Y[(mrow + r) * D + col] = acc[r];
    }
}

// ---------------------------------------------------------------------------
// Edge scatter: agg[dst] += norm[e] * hW[src]. One edge per 32-thread group,
// 4 channels per lane (float4 gather + 4 native f32 atomics, all L2-resident).
// ---------------------------------------------------------------------------
__global__ __launch_bounds__(256) void scatter_kernel(const float* __restrict__ hW,
                                                      const long long* __restrict__ src,
                                                      const long long* __restrict__ dst,
                                                      const float* __restrict__ nrm,
                                                      float* __restrict__ agg) {
    long long t = (long long)blockIdx.x * 256 + threadIdx.x;
    int e = (int)(t >> 5);
    if (e >= E_TOTAL) return;
    int lane = (int)(t & 31);

    int s, d;
    if (e < N_EDGES) { s = (int)src[e]; d = (int)dst[e]; }
    else             { s = e - N_EDGES; d = s; }

    float w = nrm[e];
    const float4 v = *(const float4*)(hW + (long long)s * D + lane * 4);
    float* ap = agg + (long long)d * D + lane * 4;
    __hip_atomic_fetch_add(ap + 0, v.x * w, __ATOMIC_RELAXED, __HIP_MEMORY_SCOPE_AGENT);
    __hip_atomic_fetch_add(ap + 1, v.y * w, __ATOMIC_RELAXED, __HIP_MEMORY_SCOPE_AGENT);
    __hip_atomic_fetch_add(ap + 2, v.z * w, __ATOMIC_RELAXED, __HIP_MEMORY_SCOPE_AGENT);
    __hip_atomic_fetch_add(ap + 3, v.w * w, __ATOMIC_RELAXED, __HIP_MEMORY_SCOPE_AGENT);
}

// ---------------------------------------------------------------------------
// out = relu(agg + b)
// ---------------------------------------------------------------------------
__global__ void bias_relu_kernel(const float* __restrict__ agg,
                                 const float* __restrict__ b,
                                 float* __restrict__ out) {
    int i = blockIdx.x * blockDim.x + threadIdx.x;
    if (i >= N_NODES * D) return;
    float v = agg[i] + b[i & (D - 1)];
    out[i] = (v > 0.0f) ? v : 0.0f;
}

// ---------------------------------------------------------------------------
extern "C" void kernel_launch(void* const* d_in, const int* in_sizes, int n_in,
                              void* d_out, int out_size, void* d_ws, size_t ws_size,
                              hipStream_t stream) {
    (void)in_sizes; (void)n_in; (void)out_size; (void)ws_size;

    const float*     x   = (const float*)d_in[0];
    const long long* ei  = (const long long*)d_in[1];   // int64 [2, E]
    const long long* src = ei;                          // edge_index[0]
    const long long* dst = ei + N_EDGES;                // edge_index[1]
    const float* w[3] = {(const float*)d_in[2], (const float*)d_in[4], (const float*)d_in[6]};
    const float* b[3] = {(const float*)d_in[3], (const float*)d_in[5], (const float*)d_in[7]};

    // Workspace layout (floats); all offsets 16B-aligned.
    float* ws   = (float*)d_ws;
    float* deg  = ws;                     // [N]
    float* dinv = deg  + N_NODES;         // [N]
    float* nrm  = dinv + N_NODES;         // [E_TOTAL]
    float* hW   = nrm  + E_TOTAL;         // [N*D]
    float* agg  = hW   + N_NODES * D;     // [N*D]
    float* hbuf = agg  + N_NODES * D;     // [N*D]

    // --- normalization coefficients (layer-invariant) ---
    zero_kernel<<<(N_NODES + 255) / 256, 256, 0, stream>>>(deg, N_NODES);
    degree_kernel<<<(E_TOTAL + 255) / 256, 256, 0, stream>>>(dst, deg);
    dinv_kernel<<<(N_NODES + 255) / 256, 256, 0, stream>>>(deg, dinv);
    norm_kernel<<<(E_TOTAL + 255) / 256, 256, 0, stream>>>(src, dst, dinv, nrm);

    // --- 3 GCN layers ---
    const float* in = x;
    for (int l = 0; l < 3; ++l) {
        float* out = (l == 2) ? (float*)d_out : hbuf;
        gemm_wmma_f16<<<N_NODES / 16, 256, 0, stream>>>(in, w[l], hW);
        zero_kernel<<<(N_NODES * D + 255) / 256, 256, 0, stream>>>(agg, N_NODES * D);
        scatter_kernel<<<((long long)E_TOTAL * 32 + 255) / 256, 256, 0, stream>>>(hW, src, dst, nrm, agg);
        bias_relu_kernel<<<(N_NODES * D + 255) / 256, 256, 0, stream>>>(agg, b[l], out);
        in = hbuf;
    }
}
